// LiteMLA_6253472383125
// MI455X (gfx1250) — matrix operand order repaired and checked by reading, not compiled, not run on hardware
//
#include <hip/hip_runtime.h>

typedef __attribute__((ext_vector_type(16))) __bf16 v16bf;
typedef __attribute__((ext_vector_type(8)))  float  v8f;

// Problem constants
#define HH   128
#define WW   128
#define NPB  16384            // pixels per batch
#define NTOT 32768            // total pixels (B=2)
#define PTILES 2048           // NTOT/16
#define CIN_ 512
#define C3I  1536             // 3*INNER
#define C2I  1024             // 2*INNER
#define COUT_ 512

// Workspace offsets (bytes)
#define OFF_QKV  ((size_t)0)                  // 2*1536*16384*4 = 201326592
#define OFF_AGG  ((size_t)201326592)          // +201326592
#define OFF_XP   ((size_t)402653184)          // bf16 B-frags of x: 16*2048*32*16*2 = 33554432
#define OFF_W1P  ((size_t)436207616)          // bf16 A-frags of w_qkv: 96*16*32*16*2 = 1572864
#define OFF_W2P  ((size_t)437780480)          // bf16 A-frags of w_proj: 32*32*32*16*2 = 1048576
#define OFF_YP   ((size_t)438829056)          // bf16 B-frags of att out: 32*2048*32*16*2 = 67108864
#define OFF_PART ((size_t)505937920)          // vk partials: 2048*72*4 = 589824
#define OFF_VK   ((size_t)506527744)          // 2*128*72*4 = 73728
#define OFF_BNM  ((size_t)506601472)          // 512*4
#define OFF_BNR  ((size_t)506603520)          // 512*4

// ---------------------------------------------------------------------------
// Pack fp32 weight matrix (O x C) into WMMA A-fragment-major bf16:
// frag index = (ot*ktiles + kt)*32 + lane ; element e -> W[ot*16 + (lane&15)][kt*32 + K]
// with K = (lane>>4)*16 + e   (contiguous k order, matches B packing)
// ---------------------------------------------------------------------------
__global__ void pack_wA(const float* __restrict__ W, __bf16* __restrict__ P,
                        int ktiles, int C, int total) {
  int t = blockIdx.x * 256 + threadIdx.x;
  if (t >= total) return;
  int lane = t & 31;
  int kt   = (t >> 5) % ktiles;
  int ot   = (t >> 5) / ktiles;
  int m    = lane & 15;
  int half = lane >> 4;
  v16bf frag;
#pragma unroll
  for (int e = 0; e < 16; ++e) {
    int K = (half << 4) + e;
    frag[e] = (__bf16)W[(size_t)(ot * 16 + m) * C + kt * 32 + K];
  }
  ((v16bf*)P)[((size_t)(ot * ktiles + kt)) * 32 + lane] = frag;
}

// ---------------------------------------------------------------------------
// Pack x (B,512,H,W) into WMMA B-fragment-major bf16 (K=channel, N=pixel):
// lane: col = lane&15 ; K = (lane>>4)*16 + e      (contiguous k order, matches A packing)
// ---------------------------------------------------------------------------
__global__ void pack_X(const float* __restrict__ x, __bf16* __restrict__ P) {
  int t = blockIdx.x * 256 + threadIdx.x;      // 16*2048*32 = 1048576
  int lane  = t & 31;
  int ptile = (t >> 5) & 2047;
  int kt    = t >> 16;
  int p  = ptile * 16 + (lane & 15);
  int b  = p >> 14, hw = p & 16383;
  int half = lane >> 4;
  v16bf frag;
#pragma unroll
  for (int e = 0; e < 16; ++e) {
    int c = kt * 32 + (half << 4) + e;
    frag[e] = (__bf16)x[((size_t)(b * CIN_ + c)) * NPB + hw];
  }
  ((v16bf*)P)[((size_t)(kt * PTILES + ptile)) * 32 + lane] = frag;
}

// ---------------------------------------------------------------------------
// Cooperative bf16 WMMA GEMM: block (8 waves) computes a 128(M) x 64(N) tile.
// B fragments are staged into LDS with GLOBAL_LOAD_ASYNC_TO_LDS_B128
// (ASYNCcnt-tracked DMA, double buffered); A fragments stay in registers and
// are reused across the 4 N-tiles. One v_wmma per (ktile, ntile).
// ---------------------------------------------------------------------------
__global__ void gemm_bf16_coop(const __bf16* __restrict__ Ap, const __bf16* __restrict__ Bp,
                               float* __restrict__ Cout, int ogroups, int ktiles, int ochan) {
  __shared__ __align__(32) char sh[2 * 4096];   // 2 buffers x (4 B-frags x 1024B)
  int tid  = threadIdx.x, lane = tid & 31, w = tid >> 5;
  int og   = blockIdx.x % ogroups;              // group of 8 otiles
  int ptg  = blockIdx.x / ogroups;              // group of 4 ptiles
  int f    = w >> 1;                            // which B fragment this wave DMAs
  int hf   = w & 1;                             // which 16B half of the 32B lane record
  int ot   = og * 8 + w;                        // this wave's otile (A row block)
  int pt0  = ptg * 4;

  unsigned ldsbase = (unsigned)(size_t)&sh[0];
  unsigned ldst    = ldsbase + (unsigned)((f << 10) + (lane << 5) + (hf << 4));
  unsigned long long gsrc0 = (unsigned long long)Bp + ((unsigned long long)(pt0 + f) << 10)
                             + ((unsigned long long)lane << 5) + ((unsigned long long)hf << 4);
  unsigned long long kstep = (unsigned long long)PTILES << 10;  // bytes per ktile of B frags

  // prologue: DMA ktile 0 into buffer 0
  {
    unsigned long long g = gsrc0;
    unsigned l = ldst;
    asm volatile("global_load_async_to_lds_b128 %0, %1, off" :: "v"(l), "v"(g) : "memory");
  }

  v8f acc0 = {}, acc1 = {}, acc2 = {}, acc3 = {};
  const v16bf* A = (const v16bf*)Ap;

  for (int kt = 0; kt < ktiles; ++kt) {
    int cur = kt & 1;
    asm volatile("s_wait_asynccnt 0x0" ::: "memory");  // own DMA complete
    __syncthreads();                                   // all waves' DMA complete
    if (kt + 1 < ktiles) {
      unsigned long long g = gsrc0 + (unsigned long long)(kt + 1) * kstep;
      unsigned l = ldst + (unsigned)((cur ^ 1) << 12);
      asm volatile("global_load_async_to_lds_b128 %0, %1, off" :: "v"(l), "v"(g) : "memory");
    }
    v16bf a = A[((size_t)(ot * ktiles + kt)) * 32 + lane];
    const char* bbuf = sh + (cur << 12) + (lane << 5);
    v16bf b0 = *(const v16bf*)(bbuf);
    v16bf b1 = *(const v16bf*)(bbuf + 1024);
    v16bf b2 = *(const v16bf*)(bbuf + 2048);
    v16bf b3 = *(const v16bf*)(bbuf + 3072);
    acc0 = __builtin_amdgcn_wmma_f32_16x16x32_bf16(false, a, false, b0, (short)0, acc0, false, false);
    acc1 = __builtin_amdgcn_wmma_f32_16x16x32_bf16(false, a, false, b1, (short)0, acc1, false, false);
    acc2 = __builtin_amdgcn_wmma_f32_16x16x32_bf16(false, a, false, b2, (short)0, acc2, false, false);
    acc3 = __builtin_amdgcn_wmma_f32_16x16x32_bf16(false, a, false, b3, (short)0, acc3, false, false);
    __syncthreads();                                   // reads done before buffer reuse
  }

  int obase = ot * 16 + (lane >> 4) * 8;
  v8f accs[4] = {acc0, acc1, acc2, acc3};
#pragma unroll
  for (int j = 0; j < 4; ++j) {
    int p  = (pt0 + j) * 16 + (lane & 15);
    int bb = p >> 14, hw = p & 16383;
#pragma unroll
    for (int r = 0; r < 8; ++r)
      Cout[((size_t)(bb * ochan + obase + r)) * NPB + hw] = accs[j][r];
  }
}

// ---------------------------------------------------------------------------
// _aggreg: 5x5 depthwise conv (pad 2) + grouped 8x8 pointwise.
// One thread per (b, group, pixel).
// ---------------------------------------------------------------------------
__global__ void aggreg_kernel(const float* __restrict__ qkv, const float* __restrict__ w_dw,
                              const float* __restrict__ w_pw, float* __restrict__ agg) {
  int t  = blockIdx.x * 256 + threadIdx.x;     // 2*192*16384 = 6291456
  int hw = t & 16383;
  int g  = (t >> 14) % 192;
  int b  = (t >> 14) / 192;
  int h  = hw >> 7, w = hw & 127;
  float dwv[8];
#pragma unroll
  for (int kc = 0; kc < 8; ++kc) {
    int c = g * 8 + kc;
    const float* src = qkv + ((size_t)(b * C3I + c)) * NPB;
    const float* wd  = w_dw + (size_t)c * 25;
    float s = 0.f;
#pragma unroll
    for (int dy = -2; dy <= 2; ++dy) {
      int hh = h + dy;
      if (hh < 0 || hh >= HH) continue;
#pragma unroll
      for (int dx = -2; dx <= 2; ++dx) {
        int ww2 = w + dx;
        if (ww2 < 0 || ww2 >= WW) continue;
        s += src[hh * WW + ww2] * wd[(dy + 2) * 5 + (dx + 2)];
      }
    }
    dwv[kc] = s;
  }
  const float* wp = w_pw + (size_t)g * 64;
#pragma unroll
  for (int o = 0; o < 8; ++o) {
    float s = 0.f;
#pragma unroll
    for (int k = 0; k < 8; ++k) s += wp[o * 8 + k] * dwv[k];
    agg[((size_t)(b * C3I + g * 8 + o)) * NPB + hw] = s;
  }
}

// ---------------------------------------------------------------------------
// Stage 1 of linear attention: vk[b,head,9,8] = sum_n [v;1] k^T (k gets relu).
// Deterministic two-stage reduction: per-block partials, then vk_reduce.
// Block = (b, head, chunk-of-2048-pixels).
// ---------------------------------------------------------------------------
__global__ void vk_accum(const float* __restrict__ qkv, const float* __restrict__ agg,
                         float* __restrict__ part) {
  int blk   = blockIdx.x;                     // 2*128*8 = 2048
  int chunk = blk & 7;
  int head  = (blk >> 3) & 127;
  int b     = blk >> 10;
  const float* src = (head < 64) ? qkv : agg;
  size_t base = ((size_t)(b * C3I + (head & 63) * 24)) * NPB;
  int tid = threadIdx.x, lane = tid & 31, wid = tid >> 5;
  float acc[9][8];
#pragma unroll
  for (int d = 0; d < 9; ++d)
#pragma unroll
    for (int e = 0; e < 8; ++e) acc[d][e] = 0.f;
  for (int i = 0; i < 8; ++i) {
    int hw = chunk * 2048 + i * 256 + tid;
    float kv[8], vv[8];
#pragma unroll
    for (int e = 0; e < 8; ++e) kv[e] = fmaxf(src[base + (size_t)(8 + e) * NPB + hw], 0.f);
#pragma unroll
    for (int d = 0; d < 8; ++d) vv[d] = src[base + (size_t)(16 + d) * NPB + hw];
#pragma unroll
    for (int d = 0; d < 8; ++d)
#pragma unroll
      for (int e = 0; e < 8; ++e) acc[d][e] += vv[d] * kv[e];
#pragma unroll
    for (int e = 0; e < 8; ++e) acc[8][e] += kv[e];
  }
  __shared__ float sh[8][72];
#pragma unroll
  for (int d = 0; d < 9; ++d)
#pragma unroll
    for (int e = 0; e < 8; ++e) {
      float v = acc[d][e];
      for (int off = 16; off > 0; off >>= 1) v += __shfl_xor(v, off, 32);
      if (lane == 0) sh[wid][d * 8 + e] = v;
    }
  __syncthreads();
  if (tid < 72) {
    float s = 0.f;
#pragma unroll
    for (int w = 0; w < 8; ++w) s += sh[w][tid];
    part[(size_t)blk * 72 + tid] = s;
  }
}

__global__ void vk_reduce(const float* __restrict__ part, float* __restrict__ vk) {
  int t = blockIdx.x * 256 + threadIdx.x;     // 2*128*72 = 18432
  if (t >= 18432) return;
  int bh = t / 72, i = t % 72;
  float s = 0.f;
#pragma unroll
  for (int c = 0; c < 8; ++c) s += part[((size_t)bh * 8 + c) * 72 + i];
  vk[t] = s;
}

// ---------------------------------------------------------------------------
// Stage 2: att = (vk @ relu(q)) / (denom + eps); write bf16 directly in the
// B-fragment-major layout of the proj GEMM (rows = out channel o2=head*8+d).
// 8 consecutive rows land in 8 consecutive e-slots of one lane -> one 16B store.
// ---------------------------------------------------------------------------
__global__ void att_pack(const float* __restrict__ qkv, const float* __restrict__ agg,
                         const float* __restrict__ vk, __bf16* __restrict__ Yp) {
  int t    = blockIdx.x * 256 + threadIdx.x;  // 2*128*16384 = 4194304
  int hw   = t & 16383;
  int head = (t >> 14) & 127;
  int b    = t >> 21;
  const float* src = (head < 64) ? qkv : agg;
  size_t base = ((size_t)(b * C3I + (head & 63) * 24)) * NPB + hw;
  float q[8];
#pragma unroll
  for (int e = 0; e < 8; ++e) q[e] = fmaxf(src[base + (size_t)e * NPB], 0.f);
  const float* vkp = vk + (size_t)(b * 128 + head) * 72;
  float att[9];
#pragma unroll
  for (int d = 0; d < 9; ++d) {
    float s = 0.f;
#pragma unroll
    for (int e = 0; e < 8; ++e) s += vkp[d * 8 + e] * q[e];
    att[d] = s;
  }
  float inv = 1.0f / (att[8] + 1e-15f);
  union { __bf16 h[8]; float4 f4; } u;
#pragma unroll
  for (int d = 0; d < 8; ++d) u.h[d] = (__bf16)(att[d] * inv);
  int p     = b * NPB + hw;
  int ktile = head >> 2;
  int r0    = (head & 3) * 8;                 // row within 32-row k-tile
  int lane  = ((r0 >> 4) << 4) | (p & 15);
  int e0    = r0 & 15;
  size_t off = (((size_t)(ktile * PTILES + (p >> 4))) * 32 + lane) * 16 + e0;
  *(float4*)(Yp + off) = u.f4;
}

// ---------------------------------------------------------------------------
// BatchNorm statistics: one block per channel, reduce over B*H*W = 32768.
// ---------------------------------------------------------------------------
__global__ void bn_stats(const float* __restrict__ proj, float* __restrict__ bnm,
                         float* __restrict__ bnr) {
  int o = blockIdx.x, tid = threadIdx.x, lane = tid & 31, wid = tid >> 5;
  float s = 0.f, s2 = 0.f;
  for (int b = 0; b < 2; ++b) {
    const float* p = proj + ((size_t)(b * COUT_ + o)) * NPB;
    for (int i = tid; i < NPB; i += 256) { float v = p[i]; s += v; s2 += v * v; }
  }
  for (int off = 16; off > 0; off >>= 1) { s += __shfl_xor(s, off, 32); s2 += __shfl_xor(s2, off, 32); }
  __shared__ float sh[16];
  if (lane == 0) { sh[wid] = s; sh[8 + wid] = s2; }
  __syncthreads();
  if (tid == 0) {
    float ts = 0.f, ts2 = 0.f;
#pragma unroll
    for (int w = 0; w < 8; ++w) { ts += sh[w]; ts2 += sh[8 + w]; }
    float mean = ts / (float)NTOT;
    float var  = ts2 / (float)NTOT - mean * mean;
    bnm[o] = mean;
    bnr[o] = rsqrtf(var + 1e-5f);
  }
}

// out = (proj - mean)*rstd*gamma + beta + x   (proj staged in d_out, same layout)
__global__ void bn_finalize(float* __restrict__ out, const float* __restrict__ x,
                            const float* __restrict__ bnm, const float* __restrict__ bnr,
                            const float* __restrict__ gamma, const float* __restrict__ beta) {
  size_t t = (size_t)blockIdx.x * 256 + threadIdx.x;   // 16777216
  int o = (int)((t >> 14) & 511);
  float v = out[t];
  out[t] = (v - bnm[o]) * bnr[o] * gamma[o] + beta[o] + x[t];
}

// ---------------------------------------------------------------------------
extern "C" void kernel_launch(void* const* d_in, const int* in_sizes, int n_in,
                              void* d_out, int out_size, void* d_ws, size_t ws_size,
                              hipStream_t stream) {
  (void)in_sizes; (void)n_in; (void)out_size; (void)ws_size;
  const float* x       = (const float*)d_in[0];
  const float* w_qkv   = (const float*)d_in[1];
  const float* w_dw    = (const float*)d_in[2];
  const float* w_pw    = (const float*)d_in[3];
  const float* w_proj  = (const float*)d_in[4];
  const float* bn_g    = (const float*)d_in[5];
  const float* bn_b    = (const float*)d_in[6];
  char* ws = (char*)d_ws;
  float*  qkv  = (float*)(ws + OFF_QKV);
  float*  agg  = (float*)(ws + OFF_AGG);
  __bf16* Xp   = (__bf16*)(ws + OFF_XP);
  __bf16* W1p  = (__bf16*)(ws + OFF_W1P);
  __bf16* W2p  = (__bf16*)(ws + OFF_W2P);
  __bf16* Yp   = (__bf16*)(ws + OFF_YP);
  float*  part = (float*)(ws + OFF_PART);
  float*  vk   = (float*)(ws + OFF_VK);
  float*  bnm  = (float*)(ws + OFF_BNM);
  float*  bnr  = (float*)(ws + OFF_BNR);
  float*  proj = (float*)d_out;   // proj output staged in d_out

  // 1) pack weights + activations into WMMA fragment layouts (fp32 -> bf16)
  pack_wA<<<192, 256, 0, stream>>>(w_qkv,  W1p, 16, 512,  96 * 16 * 32);
  pack_wA<<<128, 256, 0, stream>>>(w_proj, W2p, 32, 1024, 32 * 32 * 32);
  pack_X <<<4096, 256, 0, stream>>>(x, Xp);
  // 2) qkv GEMM (1536x512 x 32768): async-LDS staged, WMMA bf16
  //    12 ogroups (of 8 otiles) x 512 ptile-groups (of 4) = 6144 blocks
  gemm_bf16_coop<<<6144, 256, 0, stream>>>(W1p, Xp, qkv, 12, 16, C3I);
  // 3) depthwise 5x5 + grouped 8x8 pointwise
  aggreg_kernel<<<24576, 256, 0, stream>>>(qkv, w_dw, w_pw, agg);
  // 4) linear attention global reduction (deterministic two-stage)
  vk_accum <<<2048, 256, 0, stream>>>(qkv, agg, part);
  vk_reduce<<<72,   256, 0, stream>>>(part, vk);
  // 5) attention apply; emit proj-GEMM B-fragments directly
  att_pack<<<16384, 256, 0, stream>>>(qkv, agg, vk, Yp);
  // 6) proj GEMM (512x1024 x 32768): 4 ogroups x 512 ptile-groups = 2048 blocks
  gemm_bf16_coop<<<2048, 256, 0, stream>>>(W2p, Yp, proj, 4, 32, COUT_);
  // 7) BatchNorm stats + finalize (+ residual)
  bn_stats<<<512, 256, 0, stream>>>(proj, bnm, bnr);
  bn_finalize<<<65536, 256, 0, stream>>>(proj, x, bnm, bnr, bn_g, bn_b);
}
